// RegionProposoalNetwork_5686536700072
// MI455X (gfx1250) — matrix operand order repaired and checked by reading, not compiled, not run on hardware
//
#include <hip/hip_runtime.h>
#include <math.h>

// ---------------------------------------------------------------------------
// RPN forward for MI455X (gfx1250, wave32, WMMA).
//   k_wcvt      : conv_w fp32 -> bf16, layout [oc][ (kh*3+kw)*256 + ic ]
//   k_fzero     : zero padded bf16 feature buffer (1-px halo per level)
//   k_fcvt      : feats fp32 NCHW -> bf16 channel-last padded [n][pp][ic]
//   k_conv3x3   : implicit-GEMM 3x3 conv + bias + relu, v_wmma_f32_16x16x32_bf16
//                 (halo padding => branch-free, immediate-offset hot loop)
//   k_heads     : 1x1 cls/bbox heads + anchor decode + sigmoid
//   k_initcand  : init candidate slots
//   k_select    : per (image,level) top-1000 radix-select (LDS histogram)
//   k_nms       : per image LDS bitonic sort + bitmask NMS + compaction
// ---------------------------------------------------------------------------

typedef __bf16 bf16_t;
typedef __bf16 v16bf __attribute__((ext_vector_type(16)));
typedef float  v8f   __attribute__((ext_vector_type(8)));

#define TOTPOS   21824          // 128^2+64^2+32^2+16^2+8^2
#define TOTPAD   22836          // 130^2+66^2+34^2+18^2+10^2  (with halo)
#define TOTHWA   65472          // 3*TOTPOS
#define KTOT     2304           // 9*256
#define PRE_N    1000
#define POST_N   1000
#define CANDPI   4096           // padded candidates per image (3960 real)
#define NEG_INF  (-3.402823466e38f)
#define BBOX_CLAMP 4.135166556742356f  // log(1000/16)

// ---------------------------- weight convert -------------------------------
__global__ void k_wcvt(const float* __restrict__ w, bf16_t* __restrict__ wbf) {
  int i = blockIdx.x * 256 + threadIdx.x;
  if (i >= 256 * KTOT) return;
  int oc = i / KTOT, k = i - oc * KTOT;
  int q = k >> 8, ic = k & 255;
  int kh = q / 3, kw = q - 3 * kh;
  wbf[i] = (bf16_t)w[((oc * 256 + ic) * 3 + kh) * 3 + kw];
}

// ---------------------------- padded feature zero --------------------------
__global__ void k_fzero(unsigned* __restrict__ p, int ndw) {
  int i = blockIdx.x * 256 + threadIdx.x;
  if (i < ndw) p[i] = 0u;
}

// ---------------------------- feature convert ------------------------------
// f: [2][256][pos] fp32 -> fb (offset to this level's padded base):
//   fb[n][ (y+1)*Wp + (x+1) ][ic] bf16, image stride TOTPAD*256
__global__ void k_fcvt(const float* __restrict__ f, bf16_t* __restrict__ fb,
                       int pos, int W, int Wp) {
  int i = blockIdx.x * 256 + threadIdx.x;
  int tot = 2 * 256 * pos;
  if (i >= tot) return;
  int n  = i / (256 * pos);
  int r  = i - n * 256 * pos;
  int ic = r / pos;
  int p  = r - ic * pos;
  int y = p / W, x = p - y * W;
  int pp = (y + 1) * Wp + (x + 1);
  fb[((size_t)n * TOTPAD + pp) * 256 + ic] = (bf16_t)f[i];
}

// ---------------------------- 3x3 conv (WMMA) ------------------------------
// Implicit GEMM: C[oc][p] = sum_q sum_ic A[oc][q*256+ic] * Fpad[tap(q,p)][ic].
// Block = 256 threads = 8 waves; wave w owns M rows [32w,32w+32) as two 16x16
// tiles; block column tile = 16 spatial positions; K stepped by 32.
// Halo padding makes all loads unconditional with immediate offsets.
__global__ __launch_bounds__(256)
void k_conv3x3(const bf16_t* __restrict__ featbf, const bf16_t* __restrict__ wbf,
               const float* __restrict__ bias, float* __restrict__ t,
               int W, int Wp, int pos, int lvlOff, int padOff) {
  int n = blockIdx.y;
  const bf16_t* F = featbf + (size_t)(n * TOTPAD + padOff) * 256;
  float*        T = t      + (size_t)(n * TOTPOS + lvlOff) * 256;

  int lane  = threadIdx.x & 31;
  int wv    = threadIdx.x >> 5;
  int h     = lane >> 4;        // half-wave
  int l15   = lane & 15;
  int mbase = wv * 32;
  int p = blockIdx.x * 16 + l15;   // pos is always a multiple of 16
  int y = p / W, x = p - y * W;

  union { v16bf v; unsigned u[8]; } a0, a1, b;
  union { v8f v; float f[8]; } acc0, acc1;
#pragma unroll
  for (int e = 0; e < 8; ++e) { acc0.f[e] = 0.f; acc1.f[e] = 0.f; }

  // A row bases (16-bit A 16x32 layout): ka = 8h + (r<4 ? 2r : 8+2r) + icb
  const bf16_t* A0 = wbf + (size_t)(mbase + l15) * KTOT + 8 * h;
  const bf16_t* A1 = A0 + (size_t)16 * KTOT;

  for (int kh = 0; kh < 3; ++kh) {
    // per-tap feature row base; +8h selects this half-wave's K range
    const bf16_t* Frow = F + ((size_t)(y + kh) * Wp + x) * 256 + 8 * h;
#pragma unroll
    for (int kw = 0; kw < 3; ++kw) {
      const bf16_t* Fq  = Frow + kw * 256;
      const bf16_t* Aq0 = A0 + (size_t)(kh * 3 + kw) * 256;
      const bf16_t* Aq1 = A1 + (size_t)(kh * 3 + kw) * 256;
#pragma unroll
      for (int icb = 0; icb < 256; icb += 32) {
#pragma unroll
        for (int r = 0; r < 8; ++r) {
          int ao = icb + ((r < 4) ? 2 * r : 8 + 2 * r);
          a0.u[r] = *(const unsigned*)(Aq0 + ao);
          a1.u[r] = *(const unsigned*)(Aq1 + ao);
          b.u[r]  = *(const unsigned*)(Fq + ao);
        }
        acc0.v = __builtin_amdgcn_wmma_f32_16x16x32_bf16(
            false, a0.v, false, b.v, (short)0, acc0.v, false, false);
        acc1.v = __builtin_amdgcn_wmma_f32_16x16x32_bf16(
            false, a1.v, false, b.v, (short)0, acc1.v, false, false);
      }
    }
  }

#pragma unroll
  for (int e = 0; e < 8; ++e) {
    int ml = e + 8 * h;
    int ma = mbase + ml;
    int mb = mbase + 16 + ml;
    float v0 = acc0.f[e] + bias[ma];
    float v1 = acc1.f[e] + bias[mb];
    T[(size_t)ma * pos + p] = v0 > 0.f ? v0 : 0.f;
    T[(size_t)mb * pos + p] = v1 > 0.f ? v1 : 0.f;
  }
}

// ---------------------------- heads + decode -------------------------------
__global__ __launch_bounds__(256)
void k_heads(const float* __restrict__ t,
             const float* __restrict__ clsw, const float* __restrict__ clsb,
             const float* __restrict__ bw,   const float* __restrict__ bb,
             float* __restrict__ sel, float* __restrict__ sc,
             float* __restrict__ boxes,
             int W, int pos, int lvlOff, int hwaOff,
             float stride, float asize) {
  __shared__ float wsh[15 * 256];
  for (int i = threadIdx.x; i < 15 * 256; i += 256) {
    int j = i >> 8, ic = i & 255;
    wsh[i] = (j < 3) ? clsw[j * 256 + ic] : bw[(j - 3) * 256 + ic];
  }
  __syncthreads();

  int n = blockIdx.y;
  int p = blockIdx.x * 256 + threadIdx.x;
  if (p >= pos) return;
  const float* T = t + (size_t)(n * TOTPOS + lvlOff) * 256;

  float acc[15];
#pragma unroll
  for (int j = 0; j < 15; ++j) acc[j] = (j < 3) ? clsb[j] : bb[j - 3];
  for (int ic = 0; ic < 256; ++ic) {
    float v = T[(size_t)ic * pos + p];
#pragma unroll
    for (int j = 0; j < 15; ++j) acc[j] += v * wsh[j * 256 + ic];
  }

  int y = p / W, x = p - y * W;
  float gx = x * stride, gy = y * stride;
  const float ratios[3] = {0.5f, 1.0f, 2.0f};
#pragma unroll
  for (int a = 0; a < 3; ++a) {
    float sr = sqrtf(ratios[a]);
    float hs = asize * sr, ws_ = asize / sr;
    float obj = acc[a];
    float dx = acc[3 + a * 4 + 0], dy = acc[3 + a * 4 + 1];
    float dw = fminf(acc[3 + a * 4 + 2], BBOX_CLAMP);
    float dh = fminf(acc[3 + a * 4 + 3], BBOX_CLAMP);
    float px = dx * ws_ + gx, py = dy * hs + gy;
    float pw = expf(dw) * ws_, ph = expf(dh) * hs;
    float x1 = fminf(fmaxf(px - 0.5f * pw, 0.f), 512.f);
    float y1 = fminf(fmaxf(py - 0.5f * ph, 0.f), 512.f);
    float x2 = fminf(fmaxf(px + 0.5f * pw, 0.f), 512.f);
    float y2 = fminf(fmaxf(py + 0.5f * ph, 0.f), 512.f);
    bool valid = (x2 - x1 >= 1e-3f) && (y2 - y1 >= 1e-3f);
    float prob = 1.f / (1.f + expf(-obj));
    size_t idx = (size_t)n * TOTHWA + hwaOff + (size_t)p * 3 + a;
    sel[idx] = prob;                       // selection key (monotone in logit)
    sc[idx]  = valid ? prob : NEG_INF;     // final score
    boxes[idx * 4 + 0] = x1; boxes[idx * 4 + 1] = y1;
    boxes[idx * 4 + 2] = x2; boxes[idx * 4 + 3] = y2;
  }
}

// ---------------------------- candidate init -------------------------------
__global__ void k_initcand(float* cand_s, float* cand_b) {
  int i = blockIdx.x * 256 + threadIdx.x;
  if (i >= 2 * CANDPI) return;
  cand_s[i] = NEG_INF;
  cand_b[4 * i + 0] = 0.f; cand_b[4 * i + 1] = 0.f;
  cand_b[4 * i + 2] = 0.f; cand_b[4 * i + 3] = 0.f;
}

// ---------------------------- per-level top-k ------------------------------
__global__ __launch_bounds__(256)
void k_select(const float* __restrict__ sel, const float* __restrict__ sc,
              const float* __restrict__ boxes,
              float* __restrict__ cand_s, float* __restrict__ cand_b) {
  const int hwaArr[5]  = {49152, 12288, 3072, 768, 192};
  const int hwaOffA[5] = {0, 49152, 61440, 64512, 65280};
  const int candOffA[5]= {0, 1000, 2000, 3000, 3768};
  int n = blockIdx.x / 5, lvl = blockIdx.x - 5 * n;
  int HWA = hwaArr[lvl];
  int k = HWA < PRE_N ? HWA : PRE_N;
  const float* S  = sel + (size_t)n * TOTHWA + hwaOffA[lvl];
  const float* SC = sc  + (size_t)n * TOTHWA + hwaOffA[lvl];
  const float* B  = boxes + ((size_t)n * TOTHWA + hwaOffA[lvl]) * 4;
  float* CS = cand_s + (size_t)n * CANDPI + candOffA[lvl];
  float* CB = cand_b + ((size_t)n * CANDPI + candOffA[lvl]) * 4;

  if (HWA <= PRE_N) {          // take everything
    for (int i = threadIdx.x; i < HWA; i += 256) {
      CS[i] = SC[i];
      CB[4 * i + 0] = B[4 * i + 0]; CB[4 * i + 1] = B[4 * i + 1];
      CB[4 * i + 2] = B[4 * i + 2]; CB[4 * i + 3] = B[4 * i + 3];
    }
    return;
  }

  // radix-select: sel keys are sigmoid in (0,1) -> raw fp32 bits are ordered
  __shared__ unsigned hist[256];
  __shared__ unsigned sh_prefix;
  __shared__ int      sh_remaining;
  unsigned prefix = 0;
  int remaining = k;
  for (int round = 0; round < 4; ++round) {
    for (int i = threadIdx.x; i < 256; i += 256) hist[i] = 0u;
    __syncthreads();
    int sh = 24 - 8 * round;
    for (int i = threadIdx.x; i < HWA; i += 256) {
      unsigned key = __float_as_uint(S[i]);
      bool match = (round == 0) || ((key >> (sh + 8)) == prefix);
      if (match) atomicAdd(&hist[(key >> sh) & 255u], 1u);
    }
    __syncthreads();
    if (threadIdx.x == 0) {
      int cum = 0, bsel = 0;
      for (int bb_ = 255; bb_ >= 0; --bb_) {
        int c = (int)hist[bb_];
        if (cum + c >= remaining) { bsel = bb_; break; }
        cum += c;
      }
      sh_prefix = (prefix << 8) | (unsigned)bsel;
      sh_remaining = remaining - cum;
    }
    __syncthreads();
    prefix = sh_prefix;
    remaining = sh_remaining;
    __syncthreads();
  }
  unsigned T = prefix;
  int nGreater = k - remaining;

  __shared__ int cntA, cntB;
  if (threadIdx.x == 0) { cntA = 0; cntB = 0; }
  __syncthreads();
  for (int i = threadIdx.x; i < HWA; i += 256) {
    unsigned key = __float_as_uint(S[i]);
    int slot = -1;
    if (key > T) slot = atomicAdd(&cntA, 1);
    else if (key == T) {
      int e = atomicAdd(&cntB, 1);
      if (e < remaining) slot = nGreater + e;
    }
    if (slot >= 0 && slot < k) {
      CS[slot] = SC[i];
      CB[4 * slot + 0] = B[4 * i + 0]; CB[4 * slot + 1] = B[4 * i + 1];
      CB[4 * slot + 2] = B[4 * i + 2]; CB[4 * slot + 3] = B[4 * i + 3];
    }
  }
}

// ---------------------------- NMS ------------------------------------------
// dyn LDS: key[4096] f32 | pay[4096] u32 | bx[4][4096] f32 | supm[128] u32
__global__ __launch_bounds__(512)
void k_nms(const float* __restrict__ cand_s, const float* __restrict__ cand_b,
           float* __restrict__ out) {
  extern __shared__ char smem[];
  float*    key  = (float*)smem;
  unsigned* pay  = (unsigned*)(key + CANDPI);
  float*    bx   = (float*)(pay + CANDPI);           // SoA x1,y1,x2,y2
  unsigned* supm = (unsigned*)(bx + 4 * CANDPI);

  int img = blockIdx.x;
  int tid = threadIdx.x;
  const float* CS = cand_s + (size_t)img * CANDPI;
  const float* CB = cand_b + (size_t)img * CANDPI * 4;
  float* outb = out + (size_t)img * POST_N * 4;
  float* outs = out + (size_t)2 * POST_N * 4 + (size_t)img * POST_N;

  for (int i = tid; i < POST_N * 4; i += 512) outb[i] = 0.f;
  for (int i = tid; i < POST_N; i += 512) outs[i] = 0.f;
  for (int i = tid; i < CANDPI; i += 512) { key[i] = CS[i]; pay[i] = (unsigned)i; }
  for (int i = tid; i < 128; i += 512) supm[i] = 0u;
  __syncthreads();

  // bitonic sort, descending by score
  for (int kk = 2; kk <= CANDPI; kk <<= 1) {
    for (int j = kk >> 1; j > 0; j >>= 1) {
      for (int i = tid; i < CANDPI; i += 512) {
        int pr = i ^ j;
        if (pr > i) {
          bool desc = ((i & kk) == 0);
          float ki = key[i], kp = key[pr];
          if (desc ? (ki < kp) : (ki > kp)) {
            key[i] = kp; key[pr] = ki;
            unsigned tpy = pay[i]; pay[i] = pay[pr]; pay[pr] = tpy;
          }
        }
      }
      __syncthreads();
    }
  }

  // gather boxes, apply batched-NMS level offset (lvl * 2 * IMG)
  for (int i = tid; i < CANDPI; i += 512) {
    int s = (int)pay[i];
    int lvl = s < 1000 ? 0 : s < 2000 ? 1 : s < 3000 ? 2 : s < 3768 ? 3 : 4;
    float off = (float)lvl * 1024.0f;
    bx[i]              = CB[4 * s + 0] + off;
    bx[CANDPI + i]     = CB[4 * s + 1] + off;
    bx[2 * CANDPI + i] = CB[4 * s + 2] + off;
    bx[3 * CANDPI + i] = CB[4 * s + 3] + off;
  }
  __syncthreads();

  // greedy suppression: serial over i, parallel over j
  for (int i = 0; i < CANDPI; ++i) {
    bool sup_i = (supm[i >> 5] >> (i & 31)) & 1u;
    if (!sup_i && key[i] > 0.5f * NEG_INF) {
      float x1 = bx[i], y1 = bx[CANDPI + i];
      float x2 = bx[2 * CANDPI + i], y2 = bx[3 * CANDPI + i];
      float ai = (x2 - x1) * (y2 - y1);
      for (int j = i + 1 + tid; j < CANDPI; j += 512) {
        float jx1 = bx[j], jy1 = bx[CANDPI + j];
        float jx2 = bx[2 * CANDPI + j], jy2 = bx[3 * CANDPI + j];
        float w = fmaxf(fminf(x2, jx2) - fmaxf(x1, jx1), 0.f);
        float h = fmaxf(fminf(y2, jy2) - fmaxf(y1, jy1), 0.f);
        float inter = w * h;
        float aj = (jx2 - jx1) * (jy2 - jy1);
        float iou = inter / (ai + aj - inter + 1e-9f);
        if (iou > 0.7f) atomicOr(&supm[j >> 5], 1u << (j & 31));
      }
    }
    __syncthreads();
  }

  // in-order compaction of kept boxes (scores are sorted desc)
  if (tid == 0) {
    int cnt = 0;
    for (int i = 0; i < CANDPI && cnt < POST_N; ++i) {
      bool sup_i = (supm[i >> 5] >> (i & 31)) & 1u;
      if (!sup_i && key[i] > 0.5f * NEG_INF) {
        int s = (int)pay[i];
        int lvl = s < 1000 ? 0 : s < 2000 ? 1 : s < 3000 ? 2 : s < 3768 ? 3 : 4;
        float off = (float)lvl * 1024.0f;
        outb[cnt * 4 + 0] = bx[i] - off;
        outb[cnt * 4 + 1] = bx[CANDPI + i] - off;
        outb[cnt * 4 + 2] = bx[2 * CANDPI + i] - off;
        outb[cnt * 4 + 3] = bx[3 * CANDPI + i] - off;
        outs[cnt] = key[i];
        ++cnt;
      }
    }
  }
}

// ---------------------------------------------------------------------------
extern "C" void kernel_launch(void* const* d_in, const int* in_sizes, int n_in,
                              void* d_out, int out_size, void* d_ws, size_t ws_size,
                              hipStream_t stream) {
  (void)in_sizes; (void)n_in; (void)out_size; (void)ws_size;
  const float* feats[5];
  for (int i = 0; i < 5; ++i) feats[i] = (const float*)d_in[i];
  const float* conv_w = (const float*)d_in[5];
  const float* conv_b = (const float*)d_in[6];
  const float* cls_w  = (const float*)d_in[7];
  const float* cls_b  = (const float*)d_in[8];
  const float* bbox_w = (const float*)d_in[9];
  const float* bbox_b = (const float*)d_in[10];

  char* ws = (char*)d_ws;
  size_t off = 0;
  auto take = [&](size_t bytes) {
    size_t o = off;
    off = (off + bytes + 255) & ~(size_t)255;
    return o;
  };
  bf16_t* wbf    = (bf16_t*)(ws + take((size_t)256 * KTOT * 2));
  bf16_t* featbf = (bf16_t*)(ws + take((size_t)2 * TOTPAD * 256 * 2));
  float*  t      = (float*)(ws + take((size_t)2 * TOTPOS * 256 * 4));
  float*  sel    = (float*)(ws + take((size_t)2 * TOTHWA * 4));
  float*  sc     = (float*)(ws + take((size_t)2 * TOTHWA * 4));
  float*  boxes  = (float*)(ws + take((size_t)2 * TOTHWA * 4 * 4));
  float*  cand_s = (float*)(ws + take((size_t)2 * CANDPI * 4));
  float*  cand_b = (float*)(ws + take((size_t)2 * CANDPI * 4 * 4));

  const int   hwA[5]     = {128, 64, 32, 16, 8};
  const int   posA[5]    = {16384, 4096, 1024, 256, 64};
  const int   lvlOffA[5] = {0, 16384, 20480, 21504, 21760};
  const int   padOffA[5] = {0, 16900, 21256, 22412, 22736};
  const int   hwaOffA[5] = {0, 49152, 61440, 64512, 65280};
  const float strideA[5] = {4.f, 8.f, 16.f, 32.f, 64.f};
  const float sizeA[5]   = {32.f, 64.f, 128.f, 256.f, 512.f};

  k_wcvt<<<(256 * KTOT + 255) / 256, 256, 0, stream>>>(conv_w, wbf);
  {
    int ndw = TOTPAD * 256;   // 2*TOTPAD*256 bf16 = TOTPAD*256 dwords... *2 imgs
    ndw = 2 * TOTPAD * 256 / 2;     // total bytes / 4
    k_fzero<<<(ndw + 255) / 256, 256, 0, stream>>>((unsigned*)featbf, ndw);
  }
  for (int l = 0; l < 5; ++l) {
    int tot = 2 * 256 * posA[l];
    k_fcvt<<<(tot + 255) / 256, 256, 0, stream>>>(
        feats[l], featbf + (size_t)padOffA[l] * 256, posA[l],
        hwA[l], hwA[l] + 2);
  }
  for (int l = 0; l < 5; ++l) {
    dim3 g(posA[l] / 16, 2);
    k_conv3x3<<<g, 256, 0, stream>>>(featbf, wbf, conv_b, t,
                                     hwA[l], hwA[l] + 2, posA[l],
                                     lvlOffA[l], padOffA[l]);
  }
  for (int l = 0; l < 5; ++l) {
    dim3 g((posA[l] + 255) / 256, 2);
    k_heads<<<g, 256, 0, stream>>>(t, cls_w, cls_b, bbox_w, bbox_b,
                                   sel, sc, boxes,
                                   hwA[l], posA[l], lvlOffA[l],
                                   hwaOffA[l], strideA[l], sizeA[l]);
  }
  k_initcand<<<(2 * CANDPI + 255) / 256, 256, 0, stream>>>(cand_s, cand_b);
  k_select<<<10, 256, 0, stream>>>(sel, sc, boxes, cand_s, cand_b);
  size_t nms_lds = (size_t)CANDPI * 4 /*key*/ + (size_t)CANDPI * 4 /*pay*/
                 + (size_t)CANDPI * 16 /*boxes*/ + 512 /*supm*/;
  k_nms<<<2, 512, nms_lds, stream>>>(cand_s, cand_b, (float*)d_out);
}
